// MixingNetwork_48309792145472
// MI455X (gfx1250) — matrix operand order repaired and checked by reading, not compiled, not run on hardware
//
#include <hip/hip_runtime.h>

// Problem constants (from reference): B=256, S=512 -> N=131072 rows
#define NROWS 131072
#define CDIM  256   // central state dim (K of all GEMMs)
#define HDIM  256   // hypernet hidden dim
#define ADIM  8     // agents
#define EDIM  64    // embed dim
#define MTILE 128   // rows per workgroup
#define LDH   264   // padded bf16 row stride (st / h tiles)
#define LDW   264   // padded bf16 row stride (transposed weight chunk)
#define WTBUF (64 * LDW)   // one staging buffer, elements

typedef __attribute__((ext_vector_type(8)))  float  v8f;
typedef __attribute__((ext_vector_type(16))) __bf16 v16bf;

union ABu { unsigned u[8]; v16bf v; };

struct __align__(16) Smem {
  unsigned short st[MTILE * LDH];   // bf16 state tile  [r][k]
  unsigned short h [MTILE * LDH];   // bf16 elu(layer1) tile, reused per branch
  unsigned short wt[2 * WTBUF];     // double-buffered transposed weight chunk [n][k]
  float q[MTILE * ADIM];            // agent q values
  float hid[MTILE * EDIM];          // relu'd mixing hidden
  float joint[MTILE];               // partial joint value (hidden . w2)
};

__device__ inline unsigned short f2bf(float x) {
  unsigned u = __builtin_bit_cast(unsigned, x);
  unsigned r = (u + 0x7FFFu + ((u >> 16) & 1u)) >> 16;   // RNE
  return (unsigned short)r;
}
__device__ inline unsigned pack2bf(float lo, float hi) {
  return ((unsigned)f2bf(hi) << 16) | (unsigned)f2bf(lo);
}
__device__ inline float bf2f(unsigned short h) {
  unsigned u = ((unsigned)h) << 16;
  return __builtin_bit_cast(float, u);
}

// A fragment: 16x32 bf16, rows m0..m0+15, K = k0..k0+31.
// Lane L(0-15): row L, K in {0..7,16..23}; lane L+16: same row, K in {8..15,24..31}.
__device__ inline v16bf load_frag_A(const unsigned short* lds, int ld, int m0, int k0, int lane) {
  int r = lane & 15, kh = lane >> 4;
  const unsigned short* p = lds + (m0 + r) * ld + k0 + kh * 8;
  ABu x;
#pragma unroll
  for (int v = 0; v < 8; ++v)
    x.u[v] = *(const unsigned*)(p + ((v & 4) << 2) + ((v & 3) << 1));
  return x.v;
}

// B fragment: 32x16 bf16 from LDS holding W transposed ([n][k]).
// Lane L(0-15): col L, K in {0..7,16..23}; lane L+16: col L, K in {8..15,24..31}.
__device__ inline v16bf load_frag_B(const unsigned short* ldsT, int ld, int n0, int k0, int lane) {
  int c = lane & 15, kh = lane >> 4;
  const unsigned short* p = ldsT + (n0 + c) * ld + k0 + kh * 8;
  ABu x;
#pragma unroll
  for (int v = 0; v < 8; ++v)
    x.u[v] = *(const unsigned*)(p + ((v & 4) << 2) + ((v & 3) << 1));
  return x.v;
}

__device__ inline v8f wmma_bf16(v16bf a, v16bf b, v8f c) {
  return __builtin_amdgcn_wmma_f32_16x16x32_bf16(false, a, false, b, (short)0, c, false, false);
}

// ---- double-buffered weight staging: 64 rows x 256 K of bf16 = 2048 uint4
struct WtRegs { uint4 r[8]; };

__device__ inline void stage_load(WtRegs& w, const unsigned short* __restrict__ WT, int tid) {
  const uint4* src = (const uint4*)WT;
#pragma unroll
  for (int i = 0; i < 8; ++i) w.r[i] = src[tid + i * 256];
}
__device__ inline void stage_store(Smem& s, int buf, const WtRegs& w, int tid) {
  unsigned short* dst = s.wt + buf * WTBUF;
#pragma unroll
  for (int i = 0; i < 8; ++i) {
    int idx = tid + i * 256;
    int n = idx >> 5, kq = idx & 31;
    *(uint4*)&dst[n * LDW + kq * 8] = w.r[i];
  }
}

// h = elu(st @ W + bias); WT: [HDIM][CDIM] bf16 (transposed); result -> s.h (bf16)
// Weight chunks are double-buffered: global loads for chunk i+1 are issued
// before the WMMAs of chunk i, so VMEM latency hides behind matrix work.
__device__ void dense_elu_layer(Smem& s, const unsigned short* __restrict__ WT,
                                const float* __restrict__ bias,
                                int lane, int wv, int tid) {
  const int m0 = wv * 16;
  const int c = lane & 15, kh = lane >> 4;
  WtRegs regs;
  stage_load(regs, WT, tid);
  stage_store(s, 0, regs, tid);
  __syncthreads();
  for (int ch = 0; ch < 4; ++ch) {
    if (ch + 1 < 4) stage_load(regs, WT + (size_t)(ch + 1) * 64 * CDIM, tid);
    const unsigned short* wtb = s.wt + (ch & 1) * WTBUF;
    const int n0 = ch * 64;
#pragma unroll
    for (int ct = 0; ct < 4; ++ct) {
      v8f acc = {0.f, 0.f, 0.f, 0.f, 0.f, 0.f, 0.f, 0.f};
#pragma unroll
      for (int k0 = 0; k0 < CDIM; k0 += 32)
        acc = wmma_bf16(load_frag_A(s.st, LDH, m0, k0, lane),
                        load_frag_B(wtb, LDW, ct * 16, k0, lane), acc);
      int gc = n0 + ct * 16 + c;
      float bv = bias[gc];
#pragma unroll
      for (int v = 0; v < 8; ++v) {
        float x = acc[v] + bv;
        float e = x > 0.f ? x : (__expf(x) - 1.f);   // ELU(alpha=1)
        s.h[(m0 + kh * 8 + v) * LDH + gc] = f2bf(e);
      }
    }
    if (ch + 1 < 4) {
      stage_store(s, (ch + 1) & 1, regs, tid);
      __syncthreads();
    }
  }
  __syncthreads();
}

// --- prep: convert fp32 weight [K=256][N] to transposed bf16 [N][K=256] in ws
__global__ void transpose_bf16_kernel(const float* __restrict__ src,
                                      unsigned short* __restrict__ dst, int N) {
  int idx = blockIdx.x * blockDim.x + threadIdx.x;
  if (idx >= N * CDIM) return;
  int n = idx >> 8, k = idx & 255;           // writes coalesced along k
  dst[idx] = f2bf(src[k * N + n]);
}

__global__ __launch_bounds__(256, 1)
void mixing_kernel(const float* __restrict__ q,   const float* __restrict__ st,
                   const unsigned short* __restrict__ w1aT, const float* __restrict__ b1a,
                   const unsigned short* __restrict__ w1bT, const float* __restrict__ b1b,
                   const unsigned short* __restrict__ w2aT, const float* __restrict__ b2a,
                   const unsigned short* __restrict__ w2bT, const float* __restrict__ b2b,
                   const unsigned short* __restrict__ wb1T, const float* __restrict__ bb1,
                   const unsigned short* __restrict__ wb2aT,const float* __restrict__ bb2a,
                   const float* __restrict__ wb2b,          const float* __restrict__ bb2b,
                   float* __restrict__ out) {
  __shared__ Smem s;
  const int tid  = threadIdx.x;
  const int lane = tid & 31, wv = tid >> 5;
  const int row0 = blockIdx.x * MTILE;
  const int m0 = wv * 16;
  const int c = lane & 15, kh = lane >> 4;

  // ---- load tiles: q (fp32) and st (fp32 -> bf16, float4 + packed cvt)
  for (int idx = tid; idx < MTILE * ADIM; idx += 256)
    s.q[idx] = q[(size_t)row0 * ADIM + idx];
  for (int idx = tid; idx < MTILE * (CDIM / 4); idx += 256) {
    int r = idx >> 6, kq = idx & 63;
    float4 f4 = ((const float4*)(st + (size_t)(row0 + r) * CDIM))[kq];
    uint2 p;
    p.x = pack2bf(f4.x, f4.y);
    p.y = pack2bf(f4.z, f4.w);
    *(uint2*)&s.st[r * LDH + kq * 4] = p;
  }
  __syncthreads();

  // ================= Branch 0: hidden = relu(sum_a q*|h1@w1b+b1b| + st@wb1+bb1)
  dense_elu_layer(s, w1aT, b1a, lane, wv, tid);   // s.h = h1

  v8f acc[4];
  WtRegs regs;
  // seed accumulators with b1 = st @ wb1 + bb1
  stage_load(regs, wb1T, tid);
  stage_store(s, 0, regs, tid);
  __syncthreads();
#pragma unroll
  for (int g = 0; g < 4; ++g) {
    v8f f = {0.f, 0.f, 0.f, 0.f, 0.f, 0.f, 0.f, 0.f};
#pragma unroll
    for (int k0 = 0; k0 < CDIM; k0 += 32)
      f = wmma_bf16(load_frag_A(s.st, LDH, m0, k0, lane),
                    load_frag_B(s.wt, LDW, g * 16, k0, lane), f);
    float bv = bb1[g * 16 + c];
#pragma unroll
    for (int v = 0; v < 8; ++v) acc[g][v] = f[v] + bv;
  }
  __syncthreads();

  // accumulate sum_a q[n,a] * |(h1 @ w1b)[n, a*64+e] + b1b|  (double-buffered)
  stage_load(regs, w1bT, tid);
  stage_store(s, 0, regs, tid);
  __syncthreads();
  for (int a = 0; a < ADIM; ++a) {
    if (a + 1 < ADIM) stage_load(regs, w1bT + (size_t)(a + 1) * EDIM * CDIM, tid);
    const unsigned short* wtb = s.wt + (a & 1) * WTBUF;
#pragma unroll
    for (int g = 0; g < 4; ++g) {
      v8f f = {0.f, 0.f, 0.f, 0.f, 0.f, 0.f, 0.f, 0.f};
#pragma unroll
      for (int k0 = 0; k0 < CDIM; k0 += 32)
        f = wmma_bf16(load_frag_A(s.h, LDH, m0, k0, lane),
                      load_frag_B(wtb, LDW, g * 16, k0, lane), f);
      float bv = b1b[a * EDIM + g * 16 + c];
#pragma unroll
      for (int v = 0; v < 8; ++v) {
        float qv = s.q[(m0 + kh * 8 + v) * ADIM + a];
        acc[g][v] += qv * fabsf(f[v] + bv);
      }
    }
    if (a + 1 < ADIM) {
      stage_store(s, (a + 1) & 1, regs, tid);
      __syncthreads();
    }
  }
  __syncthreads();
  // relu -> s.hid
#pragma unroll
  for (int g = 0; g < 4; ++g)
#pragma unroll
    for (int v = 0; v < 8; ++v)
      s.hid[(m0 + kh * 8 + v) * EDIM + g * 16 + c] = fmaxf(acc[g][v], 0.f);
  __syncthreads();

  // ================= Branch 1: w2 = |h2@w2b+b2b|; joint_partial = hidden . w2
  dense_elu_layer(s, w2aT, b2a, lane, wv, tid);   // s.h = h2
  stage_load(regs, w2bT, tid);
  stage_store(s, 0, regs, tid);
  __syncthreads();
  float jacc[8];
#pragma unroll
  for (int v = 0; v < 8; ++v) jacc[v] = 0.f;
#pragma unroll
  for (int g = 0; g < 4; ++g) {
    v8f f = {0.f, 0.f, 0.f, 0.f, 0.f, 0.f, 0.f, 0.f};
#pragma unroll
    for (int k0 = 0; k0 < CDIM; k0 += 32)
      f = wmma_bf16(load_frag_A(s.h, LDH, m0, k0, lane),
                    load_frag_B(s.wt, LDW, g * 16, k0, lane), f);
    float bv = b2b[g * 16 + c];
#pragma unroll
    for (int v = 0; v < 8; ++v) {
      float w2v = fabsf(f[v] + bv);
      float p = s.hid[(m0 + kh * 8 + v) * EDIM + g * 16 + c] * w2v;
      // reduce across the 16 lanes holding the same row (cols 0..15)
      p += __shfl_xor(p, 1, 32);
      p += __shfl_xor(p, 2, 32);
      p += __shfl_xor(p, 4, 32);
      p += __shfl_xor(p, 8, 32);
      jacc[v] += p;
    }
  }
  if (c == 0) {   // lane 0 (rows m0..m0+7) and lane 16 (rows m0+8..m0+15)
#pragma unroll
    for (int v = 0; v < 8; ++v) s.joint[m0 + kh * 8 + v] = jacc[v];
  }
  __syncthreads();

  // ================= Branch 2: b2 = elu(st@wb2a+bb2a) @ wb2b + bb2b; emit output
  dense_elu_layer(s, wb2aT, bb2a, lane, wv, tid);  // s.h = hb  (ends with barrier)
  if (tid < MTILE) {
    float sacc = bb2b[0];
#pragma unroll 4
    for (int k = 0; k < CDIM; ++k)
      sacc += bf2f(s.h[tid * LDH + k]) * wb2b[k];
    out[(size_t)row0 + tid] = s.joint[tid] + sacc;
  }
}

extern "C" void kernel_launch(void* const* d_in, const int* in_sizes, int n_in,
                              void* d_out, int out_size, void* d_ws, size_t ws_size,
                              hipStream_t stream) {
  const float* q    = (const float*)d_in[0];
  const float* st   = (const float*)d_in[1];
  const float* w1a  = (const float*)d_in[2];
  const float* b1a  = (const float*)d_in[3];
  const float* w1b  = (const float*)d_in[4];
  const float* b1b  = (const float*)d_in[5];
  const float* w2a  = (const float*)d_in[6];
  const float* b2a  = (const float*)d_in[7];
  const float* w2b  = (const float*)d_in[8];
  const float* b2b  = (const float*)d_in[9];
  const float* wb1  = (const float*)d_in[10];
  const float* bb1  = (const float*)d_in[11];
  const float* wb2a = (const float*)d_in[12];
  const float* bb2a = (const float*)d_in[13];
  const float* wb2b = (const float*)d_in[14];
  const float* bb2b = (const float*)d_in[15];
  float* out = (float*)d_out;

  // transposed bf16 weight layout in workspace (element offsets)
  unsigned short* ws = (unsigned short*)d_ws;
  unsigned short* w1aT  = ws;                    // [256][256]
  unsigned short* w2aT  = w1aT  + 256 * CDIM;    // [256][256]
  unsigned short* wb2aT = w2aT  + 256 * CDIM;    // [256][256]
  unsigned short* w1bT  = wb2aT + 256 * CDIM;    // [512][256]
  unsigned short* wb1T  = w1bT  + 512 * CDIM;    // [64][256]
  unsigned short* w2bT  = wb1T  + 64 * CDIM;     // [64][256]

  dim3 blk(256);
  hipLaunchKernelGGL(transpose_bf16_kernel, dim3(256), blk, 0, stream, w1a,  w1aT,  256);
  hipLaunchKernelGGL(transpose_bf16_kernel, dim3(256), blk, 0, stream, w2a,  w2aT,  256);
  hipLaunchKernelGGL(transpose_bf16_kernel, dim3(256), blk, 0, stream, wb2a, wb2aT, 256);
  hipLaunchKernelGGL(transpose_bf16_kernel, dim3(512), blk, 0, stream, w1b,  w1bT,  512);
  hipLaunchKernelGGL(transpose_bf16_kernel, dim3(64),  blk, 0, stream, wb1,  wb1T,  64);
  hipLaunchKernelGGL(transpose_bf16_kernel, dim3(64),  blk, 0, stream, w2b,  w2bT,  64);

  hipLaunchKernelGGL(mixing_kernel, dim3(NROWS / MTILE), blk, 0, stream,
                     q, st, w1aT, b1a, w1bT, b1b, w2aT, b2a, w2bT, b2b,
                     wb1T, bb1, wb2aT, bb2a, wb2b, bb2b, out);
}
